// CrossAttention_42219528520106
// MI455X (gfx1250) — matrix operand order repaired and checked by reading, not compiled, not run on hardware
//
#include <hip/hip_runtime.h>
#include <hip/hip_bf16.h>

// ---------------------------------------------------------------------------
// CrossAttention on MI455X (gfx1250, wave32, WMMA).
// B=8, N=M=2048, C=512, H=4, D=128.  All GEMMs via v_wmma_f32_16x16x32_bf16.
// Attention computed transposed (S^T = KV*Q^T, O^T = V^T*P^T): softmax is
// lane-local (lane = query) and P feeds the PV WMMA straight from registers.
// LDS tiles are double-buffered; next tile is prefetched into registers
// during compute so HBM latency overlaps the WMMA pipeline.
// ---------------------------------------------------------------------------

typedef __attribute__((ext_vector_type(16))) __bf16 v16bf;
typedef __attribute__((ext_vector_type(8)))  __bf16 v8bf;
typedef __attribute__((ext_vector_type(4)))  __bf16 v4bf;
typedef __attribute__((ext_vector_type(8)))  float  v8f;
typedef __attribute__((ext_vector_type(4)))  float  f32x4;

#define CDIM   512
#define NHEAD  4
#define HDIM   128
#define SEQ    2048
#define BATCH  8
#define ROWS   (BATCH * SEQ)        // 16384
#define ATT_SCALE 0.044194173824159216f   // 1/sqrt(512)

// ---- WMMA helpers ---------------------------------------------------------

__device__ __forceinline__ v8f wmma_bf16(v16bf a, v16bf b, v8f c) {
  return __builtin_amdgcn_wmma_f32_16x16x32_bf16(
      /*neg_a=*/false, a, /*neg_b=*/false, b,
      /*c_mod=*/(short)0, c, /*reuse_a=*/false, /*reuse_b=*/false);
}

// A-operand fragment (16x32, M x K). Caller passes: row_base + (hi ? 8 : 0).
// Lane (hi=0): K 0..7 and 16..23 ; lane (hi=1): K 8..15 and 24..31.
__device__ __forceinline__ v16bf load_frag_a(const __bf16* p) {
  union { v16bf v; v8bf h[2]; } u;
  u.h[0] = *(const v8bf*)(p);
  u.h[1] = *(const v8bf*)(p + 16);
  return u.v;
}

// B-operand fragment (32x16, K x N), lane = column. Caller passes:
// col_base + (hi ? 8 : 0).  K 0..7,16..23 (hi=0) / 8..15,24..31 (hi=1).
__device__ __forceinline__ v16bf load_frag_b(const __bf16* p) {
  union { v16bf v; v8bf h[2]; } u;
  u.h[0] = *(const v8bf*)(p);
  u.h[1] = *(const v8bf*)(p + 16);
  return u.v;
}

// ---------------------------------------------------------------------------
// GEMM:  Y[rows,512] = X[rows,512] @ W[512,512] + bias
// OUT_BF16: write bf16 (Q/KV staging), else f32 (final projection).
// Block: 256 threads (8 waves). Block tile 128x128, wave tile 16x128.
// Double-buffered LDS; W tile staged transposed via packed b64 stores.
// ---------------------------------------------------------------------------
template <bool OUT_BF16>
__global__ __launch_bounds__(256)
void gemm512_kernel(const float* __restrict__ X, const float* __restrict__ W,
                    const float* __restrict__ bias,
                    __bf16* __restrict__ outb, float* __restrict__ outf) {
  __shared__ __bf16 At[2][128 * 32];   // [row][k]
  __shared__ __bf16 Bt[2][128 * 32];   // [n][k]  (W tile transposed)

  const int tid  = threadIdx.x;
  const int wave = tid >> 5;
  const int lane = tid & 31;
  const int l16  = lane & 15;
  const int hi   = lane >> 4;

  const int rowBlk = blockIdx.y * 128;
  const int colBlk = blockIdx.x * 128;

  // staging assignments
  const int ar  = tid >> 1;              // A: row 0..127
  const int ac  = (tid & 1) * 16;        // A: col 0 or 16
  const int k0t = (tid >> 5) * 4;        // W: 4 k-rows
  const int n0t = (tid & 31) * 4;        // W: 4 n-cols

  float areg[16];
  float wreg[16];

  auto preload = [&](int k0) {
    const float* asrc = X + (size_t)(rowBlk + ar) * CDIM + k0 + ac;
#pragma unroll
    for (int i = 0; i < 16; i += 4) {
      f32x4 f = *(const f32x4*)(asrc + i);
#pragma unroll
      for (int j = 0; j < 4; ++j) areg[i + j] = f[j];
    }
#pragma unroll
    for (int i = 0; i < 4; ++i) {
      f32x4 f = *(const f32x4*)(W + (size_t)(k0 + k0t + i) * CDIM + colBlk + n0t);
#pragma unroll
      for (int j = 0; j < 4; ++j) wreg[i * 4 + j] = f[j];
    }
  };

  v8f acc[8];
#pragma unroll
  for (int t = 0; t < 8; ++t)
#pragma unroll
    for (int r = 0; r < 8; ++r) acc[t][r] = 0.0f;

  preload(0);

  for (int k0 = 0, it = 0; k0 < CDIM; k0 += 32, ++it) {
    const int buf = it & 1;
    // ---- store staged registers to LDS (packed) ----
    {
      __bf16* da = &At[buf][ar * 32 + ac];
#pragma unroll
      for (int half = 0; half < 2; ++half) {
        v8bf x;
#pragma unroll
        for (int j = 0; j < 8; ++j) x[j] = (__bf16)areg[half * 8 + j];
        *(v8bf*)(da + half * 8) = x;
      }
#pragma unroll
      for (int j = 0; j < 4; ++j) {      // Bt[n0t+j][k0t..k0t+3]
        v4bf t;
#pragma unroll
        for (int i = 0; i < 4; ++i) t[i] = (__bf16)wreg[i * 4 + j];
        *(v4bf*)(&Bt[buf][(n0t + j) * 32 + k0t]) = t;
      }
    }
    __syncthreads();

    if (k0 + 32 < CDIM) preload(k0 + 32);   // overlap next tile with compute

    // ---- compute: each wave does a 16x128 strip ----
    const v16bf af = load_frag_a(&At[buf][(wave * 16 + l16) * 32 + hi * 8]);
#pragma unroll
    for (int t = 0; t < 8; ++t) {
      const v16bf bf = load_frag_b(&Bt[buf][(t * 16 + l16) * 32 + hi * 8]);
      acc[t] = wmma_bf16(af, bf, acc[t]);
    }
  }

  // ---- epilogue: bias + store ----
  const int orow0 = rowBlk + wave * 16 + hi * 8;
#pragma unroll
  for (int t = 0; t < 8; ++t) {
    const int col = colBlk + t * 16 + l16;
    const float bv = bias[col];
#pragma unroll
    for (int r = 0; r < 8; ++r) {
      const float v = acc[t][r] + bv;
      const size_t idx = (size_t)(orow0 + r) * CDIM + col;
      if (OUT_BF16) outb[idx] = (__bf16)v;
      else          outf[idx] = v;
    }
  }
}

// ---------------------------------------------------------------------------
// Flash attention (transposed).  Q, KV: bf16 [16384, 512].  Xout: f32.
// grid.x = N/128 (query strip), grid.y = B*H.  Block 256 = 8 waves.
// Each wave: 16 queries (lane = query), full D=128 accumulated transposed.
//   S^T(m,n) = KV(m,d) @ Q^T(d,n)     -- A from kvr (row-major), B from Q rows
//   O^T(d,n) = V^T(d,m) @ P^T(m,n)    -- A from kvt, B = P^T straight from regs
// ---------------------------------------------------------------------------
__global__ __launch_bounds__(256)
void attn_kernel(const __bf16* __restrict__ Q, const __bf16* __restrict__ KV,
                 float* __restrict__ Xout) {
  __shared__ __bf16 kvr[2][32 * 128];     // KV tile row-major   [m][d]
  __shared__ __bf16 kvt[2][128 * 32];     // KV tile transposed  [d][m]

  const int tid  = threadIdx.x;
  const int wave = tid >> 5;
  const int lane = tid & 31;
  const int l16  = lane & 15;
  const int hi   = lane >> 4;

  const int b   = blockIdx.y >> 2;
  const int h   = blockIdx.y & 3;
  const int col0   = h * HDIM;
  const int qrow0  = b * SEQ + blockIdx.x * 128;
  const int kvrow0 = b * SEQ;

  // staging assignment: 4(m) x 4(d) sub-tile per thread
  const int m0t = (tid >> 5) * 4;        // 0..28
  const int d0t = (tid & 31) * 4;        // 0..124

  v4bf kreg[4];
  auto preload = [&](int m0) {
#pragma unroll
    for (int i = 0; i < 4; ++i)
      kreg[i] = *(const v4bf*)(KV + (size_t)(kvrow0 + m0 + m0t + i) * CDIM + col0 + d0t);
  };

  // ---- this lane's query row; Q strip as B-operand fragments (4 k-steps) --
  const int nq = qrow0 + wave * 16 + l16;
  v16bf qf[4];
  {
    const __bf16* qp = Q + (size_t)nq * CDIM + col0 + hi * 8;
#pragma unroll
    for (int kk = 0; kk < 4; ++kk) qf[kk] = load_frag_b(qp + kk * 32);
  }

  v8f o[8];                       // O^T: 8 d-tiles (lane = query)
#pragma unroll
  for (int t = 0; t < 8; ++t)
#pragma unroll
    for (int r = 0; r < 8; ++r) o[t][r] = 0.0f;
  float rmax = -1e30f, rsum = 0.0f;   // per-lane (per-query) stats, raw units

  preload(0);

  for (int m0 = 0, it = 0; m0 < SEQ; m0 += 32, ++it) {
    const int buf = it & 1;
    // ---- store staged KV registers: row-major + packed transpose ----
    {
#pragma unroll
      for (int i = 0; i < 4; ++i)
        *(v4bf*)(&kvr[buf][(m0t + i) * 128 + d0t]) = kreg[i];
#pragma unroll
      for (int j = 0; j < 4; ++j) {     // kvt[d0t+j][m0t..m0t+3]
        v4bf t;
#pragma unroll
        for (int i = 0; i < 4; ++i) t[i] = kreg[i][j];
        *(v4bf*)(&kvt[buf][(d0t + j) * 32 + m0t]) = t;
      }
    }
    __syncthreads();

    if (m0 + 32 < SEQ) preload(m0 + 32);   // overlap next tile with compute

    // ---- S^T = KV @ Q^T : two 16(m)x16(n) tiles (raw logits) ----
    v8f st[2];
#pragma unroll
    for (int j = 0; j < 2; ++j) {
#pragma unroll
      for (int r = 0; r < 8; ++r) st[j][r] = 0.0f;
#pragma unroll
      for (int kk = 0; kk < 4; ++kk) {
        const v16bf af = load_frag_a(&kvr[buf][(j * 16 + l16) * 128 + kk * 32 + hi * 8]);
        st[j] = wmma_bf16(af, qf[kk], st[j]);
      }
    }

    // ---- lane-local online softmax over the 16 m-values ----
    float mx = st[0][0];
#pragma unroll
    for (int r = 1; r < 8; ++r) mx = fmaxf(mx, st[0][r]);
#pragma unroll
    for (int r = 0; r < 8; ++r) mx = fmaxf(mx, st[1][r]);
    const float nm = fmaxf(rmax, mx);
    const float sc = __expf((rmax - nm) * ATT_SCALE);
    const float cc = -nm * ATT_SCALE;
    rmax = nm;
#pragma unroll
    for (int j = 0; j < 2; ++j)
#pragma unroll
      for (int r = 0; r < 8; ++r) st[j][r] = __expf(fmaf(st[j][r], ATT_SCALE, cc));
    float ts = 0.0f;
#pragma unroll
    for (int j = 0; j < 2; ++j)
#pragma unroll
      for (int r = 0; r < 8; ++r) ts += st[j][r];
    rsum = rsum * sc + ts;
#pragma unroll
    for (int t = 0; t < 8; ++t)
#pragma unroll
      for (int r = 0; r < 8; ++r) o[t][r] *= sc;

    // ---- P^T B-fragment straight from registers (layout-identical) ----
    v16bf pf;
#pragma unroll
    for (int r = 0; r < 8; ++r) {
      pf[r]     = (__bf16)st[0][r];
      pf[r + 8] = (__bf16)st[1][r];
    }

    // ---- O^T += V^T @ P^T ----
#pragma unroll
    for (int t = 0; t < 8; ++t) {
      const v16bf vtf = load_frag_a(&kvt[buf][(t * 16 + l16) * 32 + hi * 8]);
      o[t] = wmma_bf16(vtf, pf, o[t]);
    }
  }

  // ---- normalize and store: per lane, 8 contiguous f32 per tile ----
  const float rs = 1.0f / rsum;
#pragma unroll
  for (int t = 0; t < 8; ++t) {
    float* dst = Xout + (size_t)nq * CDIM + col0 + t * 16 + hi * 8;
    f32x4 lo, hi4;
#pragma unroll
    for (int r = 0; r < 4; ++r) { lo[r] = o[t][r] * rs; hi4[r] = o[t][r + 4] * rs; }
    *(f32x4*)(dst)     = lo;
    *(f32x4*)(dst + 4) = hi4;
  }
}

// ---------------------------------------------------------------------------
// Launch
// ---------------------------------------------------------------------------
extern "C" void kernel_launch(void* const* d_in, const int* in_sizes, int n_in,
                              void* d_out, int out_size, void* d_ws, size_t ws_size,
                              hipStream_t stream) {
  const float* F1     = (const float*)d_in[0];
  const float* F2     = (const float*)d_in[1];
  const float* W_qkv  = (const float*)d_in[2];
  const float* b_qkv  = (const float*)d_in[3];
  const float* W_proj = (const float*)d_in[4];
  const float* b_proj = (const float*)d_in[5];
  float* out = (float*)d_out;

  // workspace: Q bf16 (16MB) | KV bf16 (16MB) | X f32 (32MB)
  __bf16* Qb  = (__bf16*)d_ws;
  __bf16* KVb = Qb + (size_t)ROWS * CDIM;
  float*  Xa  = (float*)(KVb + (size_t)ROWS * CDIM);

  const dim3 gGemm(CDIM / 128, ROWS / 128);   // 4 x 128
  const dim3 gAttn(SEQ / 128, BATCH * NHEAD); // 16 x 32
  const dim3 blk(256);

  gemm512_kernel<true ><<<gGemm, blk, 0, stream>>>(F1, W_qkv, b_qkv, Qb,  nullptr);
  gemm512_kernel<true ><<<gGemm, blk, 0, stream>>>(F2, W_qkv, b_qkv, KVb, nullptr);
  attn_kernel           <<<gAttn, blk, 0, stream>>>(Qb, KVb, Xa);
  gemm512_kernel<false><<<gGemm, blk, 0, stream>>>(Xa, W_proj, b_proj, nullptr, out);
}
